// TopDown_42374147342403
// MI455X (gfx1250) — matrix operand order repaired and checked by reading, not compiled, not run on hardware
//
#include <hip/hip_runtime.h>
#include <hip/hip_bf16.h>

typedef __attribute__((ext_vector_type(16))) __bf16 v16bf;
typedef __attribute__((ext_vector_type(8)))  float  v8f;

#define BRR   768
#define RR    6
#define HH    512
#define NNN   196
#define VV    2001
#define MROWS (BRR*NNN)   // 150528 = 9408*16

// ---------------- small utility kernels ----------------
__global__ void fill_zero_f32(float* __restrict__ p, int n) {
  int i = blockIdx.x * blockDim.x + threadIdx.x;
  if (i < n) p[i] = 0.f;
}
__global__ void copy_f32(const float* __restrict__ a, float* __restrict__ b, int n) {
  int i = blockIdx.x * blockDim.x + threadIdx.x;
  if (i < n) b[i] = a[i];
}
// W[K][N] f32 (row-major, as in reference x@W) -> Wt[N][Kpad] bf16, zero-padded K tail
__global__ void prep_wt(const float* __restrict__ W, __bf16* __restrict__ Wt,
                        int K, int N, int Kpad) {
  int i = blockIdx.x * blockDim.x + threadIdx.x;
  if (i >= N * Kpad) return;
  int n = i / Kpad, k = i % Kpad;
  Wt[i] = (k < K) ? (__bf16)W[(size_t)k * N + n] : (__bf16)0.f;
}
// zero-pad each row: src[rows][K] -> dst[rows][Kpad]
__global__ void pad_rows(const float* __restrict__ src, float* __restrict__ dst,
                         int rows, int K, int Kpad) {
  int i = blockIdx.x * blockDim.x + threadIdx.x;
  if (i >= rows * Kpad) return;
  int r = i / Kpad, k = i % Kpad;
  dst[i] = (k < K) ? src[(size_t)r * K + k] : 0.f;
}

// ---------------- WMMA wave-tile GEMM core (16 x 64 per wave) ----------------
// Branch-free hot loop: M is always a multiple of 16, K pre-padded to Kpad (x32).
// A: f32 row-major [.., lda], streamed & converted to bf16 in registers.
// Bt: bf16 [N][Kpad] pre-transposed weights (L2-resident).
// Fragment layouts used by this code (wave32; A and B share one contiguous k order):
//   A 16x32 bf16: lane l, half=l>>4: elems 0..15 => K=kb+half*16+e (contiguous)
//   B 32x16 bf16: N = nbase + (l&15); elems 0..15 => K = kb + half*16 + e (contiguous 32B)
template <bool GUARDN>
__device__ __forceinline__ void wave_tile_gemm(
    const float* __restrict__ A, int lda, int Kpad,
    const __bf16* __restrict__ Bt, int N,
    int mbase, int nbase, v8f acc[4])
{
  const int lane = threadIdx.x & 31;
  const int l15  = lane & 15;
  const int half = lane >> 4;
  const float* __restrict__ Arow = A + (size_t)(mbase + l15) * lda;

  for (int kb = 0; kb < Kpad; kb += 32) {
    const int k0 = kb + half * 16;
    const float4 x0 = *(const float4*)(Arow + k0);
    const float4 x1 = *(const float4*)(Arow + k0 + 4);
    const float4 y0 = *(const float4*)(Arow + k0 + 8);
    const float4 y1 = *(const float4*)(Arow + k0 + 12);
    __builtin_prefetch(Arow + k0 + 32, 0, 1);

    v16bf af;
    af[0]=(__bf16)x0.x;  af[1]=(__bf16)x0.y;  af[2]=(__bf16)x0.z;  af[3]=(__bf16)x0.w;
    af[4]=(__bf16)x1.x;  af[5]=(__bf16)x1.y;  af[6]=(__bf16)x1.z;  af[7]=(__bf16)x1.w;
    af[8]=(__bf16)y0.x;  af[9]=(__bf16)y0.y;  af[10]=(__bf16)y0.z; af[11]=(__bf16)y0.w;
    af[12]=(__bf16)y1.x; af[13]=(__bf16)y1.y; af[14]=(__bf16)y1.z; af[15]=(__bf16)y1.w;

    #pragma unroll
    for (int t = 0; t < 4; ++t) {
      const int n = nbase + t * 16 + l15;
      v16bf bf;
      if (!GUARDN || (n < N)) {
        bf = *(const v16bf*)(Bt + (size_t)n * Kpad + kb + half * 16);
      } else {
        #pragma unroll
        for (int e = 0; e < 16; ++e) bf[e] = (__bf16)0.f;
      }
      acc[t] = __builtin_amdgcn_wmma_f32_16x16x32_bf16(
          false, af, false, bf, (short)0, acc[t], false, false);
    }
  }
}

__device__ __forceinline__ void zero_acc(v8f acc[4]) {
  #pragma unroll
  for (int t = 0; t < 4; ++t)
    #pragma unroll
    for (int v = 0; v < 8; ++v) acc[t][v] = 0.f;
}

// ---------------- GEMM with bias (+optional relu) epilogue ----------------
template <bool GUARDN>
__global__ __launch_bounds__(256) void gemm_bias_act(
    const float* __restrict__ A, int lda, int M, int Kpad,
    const __bf16* __restrict__ Bt, int N,
    const float* __restrict__ bias,
    float* __restrict__ C, int ldc, int relu)
{
  const int wave  = threadIdx.x >> 5;
  const int mbase = blockIdx.x * 16;
  const int nbase = blockIdx.y * 512 + wave * 64;
  v8f acc[4];
  zero_acc(acc);
  wave_tile_gemm<GUARDN>(A, lda, Kpad, Bt, N, mbase, nbase, acc);

  const int lane = threadIdx.x & 31, l15 = lane & 15, half = lane >> 4;
  #pragma unroll
  for (int t = 0; t < 4; ++t) {
    const int n = nbase + t * 16 + l15;
    if (GUARDN && n >= N) continue;
    const float b = bias ? bias[n] : 0.f;
    #pragma unroll
    for (int v = 0; v < 8; ++v) {
      const int m = mbase + v + 8 * half;
      float x = acc[t][v] + b;
      if (relu) x = fmaxf(x, 0.f);
      C[(size_t)m * ldc + n] = x;
    }
  }
}

// ---------------- big pass: img_org @ Wa_top fused with score epilogue ----------------
// s_img[m] = relu(acc[m,:] + qc[m/196,:]) . Wl + bl   (N=K=512, M=150528)
// A 16-row tile straddles at most one sample boundary (16 < 196), so br is a
// scalar division + one per-element compare (no per-lane magic-number divides).
__global__ __launch_bounds__(256) void gemm_img_score(
    const float* __restrict__ img,
    const __bf16* __restrict__ Wt_atop,
    const float* __restrict__ qc,
    const float* __restrict__ Wl,
    const float* __restrict__ bl,
    float* __restrict__ s_img)
{
  __shared__ float red[8][16];
  const int wave  = threadIdx.x >> 5;
  const int mbase = blockIdx.x * 16;
  const int nbase = wave * 64;
  v8f acc[4];
  zero_acc(acc);
  wave_tile_gemm<false>(img, HH, HH, Wt_atop, HH, mbase, nbase, acc);

  const int lane = threadIdx.x & 31, l15 = lane & 15, half = lane >> 4;
  const int br0 = mbase / NNN;                 // uniform -> scalar division
  const int boundary = (br0 + 1) * NNN;
  const int br1 = (br0 + 1 < BRR) ? (br0 + 1) : (BRR - 1);

  float sacc[8];
  #pragma unroll
  for (int v = 0; v < 8; ++v) sacc[v] = 0.f;

  #pragma unroll
  for (int t = 0; t < 4; ++t) {
    const int n = nbase + t * 16 + l15;
    const float wl  = Wl[n];
    const float qc0 = qc[(size_t)br0 * HH + n];
    const float qc1 = qc[(size_t)br1 * HH + n];
    #pragma unroll
    for (int v = 0; v < 8; ++v) {
      const int m = mbase + v + 8 * half;
      const float qcv = (m >= boundary) ? qc1 : qc0;
      const float h = fmaxf(acc[t][v] + qcv, 0.f);
      sacc[v] += h * wl;
    }
  }
  // deterministic reduce across the 16-lane column group (bits 0..3)
  #pragma unroll
  for (int off = 1; off < 16; off <<= 1)
    #pragma unroll
    for (int v = 0; v < 8; ++v) sacc[v] += __shfl_xor(sacc[v], off, 32);

  if (l15 == 0)
    #pragma unroll
    for (int v = 0; v < 8; ++v) red[wave][v + 8 * half] = sacc[v];
  __syncthreads();
  if (threadIdx.x < 16) {
    float s = bl[0];
    #pragma unroll
    for (int w = 0; w < 8; ++w) s += red[w][threadIdx.x];
    s_img[mbase + threadIdx.x] = s;
  }
}

// ---------------- per-sample softmax precompute over the 196 image objects ----------------
__global__ __launch_bounds__(256) void softmax_u(
    const float* __restrict__ s_img, const float* __restrict__ img,
    float* __restrict__ m_img, float* __restrict__ z_img, float* __restrict__ U)
{
  const int br = blockIdx.x;
  __shared__ float e[NNN];
  __shared__ float red[256];
  const float* s = s_img + (size_t)br * NNN;

  float lm = -3.0e38f;
  for (int j = threadIdx.x; j < NNN; j += 256) lm = fmaxf(lm, s[j]);
  red[threadIdx.x] = lm; __syncthreads();
  for (int o = 128; o > 0; o >>= 1) {
    if (threadIdx.x < o) red[threadIdx.x] = fmaxf(red[threadIdx.x], red[threadIdx.x + o]);
    __syncthreads();
  }
  float m = red[0]; __syncthreads();

  float lz = 0.f;
  for (int j = threadIdx.x; j < NNN; j += 256) { float ee = __expf(s[j] - m); e[j] = ee; lz += ee; }
  red[threadIdx.x] = lz; __syncthreads();
  for (int o = 128; o > 0; o >>= 1) {
    if (threadIdx.x < o) red[threadIdx.x] += red[threadIdx.x + o];
    __syncthreads();
  }
  if (threadIdx.x == 0) { m_img[br] = m; z_img[br] = red[0]; }
  __syncthreads();

  const float* ib = img + (size_t)br * NNN * HH;
  for (int c = threadIdx.x; c < HH; c += 256) {
    float u = 0.f;
    for (int j = 0; j < NNN; ++j) u += e[j] * ib[(size_t)j * HH + c];
    U[(size_t)br * HH + c] = u;
  }
}

// ---------------- per-iteration: leave-one-out role attention logits ----------------
__global__ __launch_bounds__(256) void role_scores(
    const float* __restrict__ Pj, const float* __restrict__ qc,
    const float* __restrict__ Wl, const float* __restrict__ bl,
    float* __restrict__ s_l)
{
  const int br = blockIdx.x;
  const int smp = br / RR, r = br % RR;
  __shared__ float red[256];
  const float* qcb = qc + (size_t)br * HH;
  for (int i = 0; i < RR - 1; ++i) {
    const int j = (i < r) ? i : i + 1;
    const float* p = Pj + ((size_t)smp * RR + j) * HH;
    float a = 0.f;
    for (int c = threadIdx.x; c < HH; c += 256) a += fmaxf(p[c] + qcb[c], 0.f) * Wl[c];
    red[threadIdx.x] = a; __syncthreads();
    for (int o = 128; o > 0; o >>= 1) {
      if (threadIdx.x < o) red[threadIdx.x] += red[threadIdx.x + o];
      __syncthreads();
    }
    if (threadIdx.x == 0) s_l[(size_t)br * (RR - 1) + i] = red[0] + bl[0];
    __syncthreads();
  }
}

// ---------------- per-iteration: merge image softmax state with role logits -> v_emb ----------------
__global__ __launch_bounds__(256) void combine_vemb(
    const float* __restrict__ U, const float* __restrict__ z_img,
    const float* __restrict__ m_img, const float* __restrict__ s_l,
    const float* __restrict__ joint, float* __restrict__ v_emb)
{
  const int br = blockIdx.x;
  const int smp = br / RR, r = br % RR;
  float sl[RR - 1];
  #pragma unroll
  for (int i = 0; i < RR - 1; ++i) sl[i] = s_l[(size_t)br * (RR - 1) + i];
  const float mi = m_img[br];
  float m = mi;
  #pragma unroll
  for (int i = 0; i < RR - 1; ++i) m = fmaxf(m, sl[i]);
  const float a = __expf(mi - m);
  float Z = z_img[br] * a;
  float e[RR - 1];
  #pragma unroll
  for (int i = 0; i < RR - 1; ++i) { e[i] = __expf(sl[i] - m); Z += e[i]; }
  const float inv = 1.f / Z;
  for (int c = threadIdx.x; c < HH; c += 256) {
    float v = U[(size_t)br * HH + c] * a;
    #pragma unroll
    for (int i = 0; i < RR - 1; ++i) {
      const int j = (i < r) ? i : i + 1;
      v += e[i] * joint[((size_t)smp * RR + j) * HH + c];
    }
    v_emb[(size_t)br * HH + c] = v * inv;
  }
}

// ---------------- per-iteration: v_repr GEMM fused with joint += q_repr * v_repr ----------------
__global__ __launch_bounds__(256) void gemm_joint_update(
    const float* __restrict__ A /*v_emb*/,
    const __bf16* __restrict__ Bt /*Wt_vn*/,
    const float* __restrict__ bvn,
    const float* __restrict__ q_repr,
    float* __restrict__ joint)
{
  const int wave  = threadIdx.x >> 5;
  const int mbase = blockIdx.x * 16;
  const int nbase = wave * 64;
  v8f acc[4];
  zero_acc(acc);
  wave_tile_gemm<false>(A, HH, HH, Bt, HH, mbase, nbase, acc);

  const int lane = threadIdx.x & 31, l15 = lane & 15, half = lane >> 4;
  #pragma unroll
  for (int t = 0; t < 4; ++t) {
    const int n = nbase + t * 16 + l15;
    const float b = bvn[n];
    #pragma unroll
    for (int v = 0; v < 8; ++v) {
      const int m = mbase + v + 8 * half;
      const float vr = fmaxf(acc[t][v] + b, 0.f);
      const size_t o = (size_t)m * HH + n;
      joint[o] += q_repr[o] * vr;
    }
  }
}

// ---------------- host orchestration ----------------
extern "C" void kernel_launch(void* const* d_in, const int* in_sizes, int n_in,
                              void* d_out, int out_size, void* d_ws, size_t ws_size,
                              hipStream_t stream)
{
  (void)in_sizes; (void)n_in; (void)out_size; (void)ws_size;
  const float* img = (const float*)d_in[0];
  const float* q   = (const float*)d_in[1];
  const float* Wq  = (const float*)d_in[2];
  const float* bq  = (const float*)d_in[3];
  const float* Wa  = (const float*)d_in[4];
  const float* ba  = (const float*)d_in[5];
  const float* Wl  = (const float*)d_in[6];
  const float* bl  = (const float*)d_in[7];
  const float* Wqn = (const float*)d_in[8];
  const float* bqn = (const float*)d_in[9];
  const float* Wvn = (const float*)d_in[10];
  const float* bvn = (const float*)d_in[11];
  const float* Wc1 = (const float*)d_in[12];
  const float* bc1 = (const float*)d_in[13];
  const float* Wc2 = (const float*)d_in[14];
  const float* bc2 = (const float*)d_in[15];
  float* out = (float*)d_out;

  char* ws = (char*)d_ws;
  size_t off = 0;
  auto alloc = [&](size_t bytes) -> char* {
    char* p = ws + off;
    off += (bytes + 255) & ~(size_t)255;
    return p;
  };
  const int KQ = 600, KQP = 608;
  __bf16* Wt_q    = (__bf16*)alloc((size_t)HH * KQP * 2);
  __bf16* Wt_atop = (__bf16*)alloc((size_t)HH * HH * 2);
  __bf16* Wt_abot = (__bf16*)alloc((size_t)HH * HH * 2);
  __bf16* Wt_qn   = (__bf16*)alloc((size_t)HH * HH * 2);
  __bf16* Wt_vn   = (__bf16*)alloc((size_t)HH * HH * 2);
  __bf16* Wt_c1   = (__bf16*)alloc((size_t)1024 * HH * 2);
  __bf16* Wt_c2   = (__bf16*)alloc((size_t)VV * 1024 * 2);
  float* q_pad  = (float*)alloc((size_t)BRR * KQP * 4);
  float* q_emb  = (float*)alloc((size_t)BRR * HH * 4);
  float* qc     = (float*)alloc((size_t)BRR * HH * 4);
  float* q_repr = (float*)alloc((size_t)BRR * HH * 4);
  float* joint  = (float*)alloc((size_t)BRR * HH * 4);
  float* Pj     = (float*)alloc((size_t)BRR * HH * 4);
  float* v_emb  = (float*)alloc((size_t)BRR * HH * 4);
  float* U      = (float*)alloc((size_t)BRR * HH * 4);
  float* s_img  = (float*)alloc((size_t)MROWS * 4);
  float* m_img  = (float*)alloc((size_t)BRR * 4);
  float* z_img  = (float*)alloc((size_t)BRR * 4);
  float* s_l    = (float*)alloc((size_t)BRR * (RR - 1) * 4);
  float* t_hid  = (float*)alloc((size_t)BRR * 1024 * 4);

  const dim3 blk(256);
  auto cdiv = [](int a, int b) { return (a + b - 1) / b; };

  // --- prep: transpose/convert weights to bf16 [N][Kpad]; zero-pad q rows ---
  prep_wt<<<cdiv(HH * KQP, 256), blk, 0, stream>>>(Wq, Wt_q, KQ, HH, KQP);
  prep_wt<<<cdiv(HH * HH, 256), blk, 0, stream>>>(Wa, Wt_atop, HH, HH, HH);
  prep_wt<<<cdiv(HH * HH, 256), blk, 0, stream>>>(Wa + (size_t)HH * HH, Wt_abot, HH, HH, HH);
  prep_wt<<<cdiv(HH * HH, 256), blk, 0, stream>>>(Wqn, Wt_qn, HH, HH, HH);
  prep_wt<<<cdiv(HH * HH, 256), blk, 0, stream>>>(Wvn, Wt_vn, HH, HH, HH);
  prep_wt<<<cdiv(1024 * HH, 256), blk, 0, stream>>>(Wc1, Wt_c1, HH, 1024, HH);
  prep_wt<<<cdiv(VV * 1024, 256), blk, 0, stream>>>(Wc2, Wt_c2, 1024, VV, 1024);
  pad_rows<<<cdiv(BRR * KQP, 256), blk, 0, stream>>>(q, q_pad, BRR, KQ, KQP);
  fill_zero_f32<<<cdiv(BRR * HH, 256), blk, 0, stream>>>(joint, BRR * HH);

  // --- loop-invariant q-side GEMMs ---
  gemm_bias_act<false><<<dim3(BRR / 16, 1), blk, 0, stream>>>(q_pad, KQP, BRR, KQP, Wt_q, HH, bq, q_emb, HH, 1);
  gemm_bias_act<false><<<dim3(BRR / 16, 1), blk, 0, stream>>>(q_emb, HH, BRR, HH, Wt_abot, HH, ba, qc, HH, 0);
  gemm_bias_act<false><<<dim3(BRR / 16, 1), blk, 0, stream>>>(q_emb, HH, BRR, HH, Wt_qn, HH, bqn, q_repr, HH, 1);

  // --- loop-invariant image pass: scores, then softmax state (m, Z, U) ---
  gemm_img_score<<<dim3(MROWS / 16, 1), blk, 0, stream>>>(img, Wt_atop, qc, Wl, bl, s_img);
  softmax_u<<<dim3(BRR), blk, 0, stream>>>(s_img, img, m_img, z_img, U);

  // --- 3 message-passing iterations (only the 5 LOO joint rows change) ---
  for (int it = 0; it < 3; ++it) {
    gemm_bias_act<false><<<dim3(BRR / 16, 1), blk, 0, stream>>>(joint, HH, BRR, HH, Wt_atop, HH, nullptr, Pj, HH, 0);
    role_scores<<<dim3(BRR), blk, 0, stream>>>(Pj, qc, Wl, bl, s_l);
    combine_vemb<<<dim3(BRR), blk, 0, stream>>>(U, z_img, m_img, s_l, joint, v_emb);
    gemm_joint_update<<<dim3(BRR / 16, 1), blk, 0, stream>>>(v_emb, Wt_vn, bvn, q_repr, joint);
  }

  // --- classifier ---
  gemm_bias_act<false><<<dim3(BRR / 16, 2), blk, 0, stream>>>(joint, HH, BRR, HH, Wt_c1, 1024, bc1, t_hid, 1024, 1);
  gemm_bias_act<true><<<dim3(BRR / 16, 4), blk, 0, stream>>>(t_hid, 1024, BRR, 1024, Wt_c2, VV, bc2, out, VV, 0);

  // outputs: (logits [768,2001], joint [768,512]) concatenated flat
  copy_f32<<<cdiv(BRR * HH, 256), blk, 0, stream>>>(joint, out + (size_t)BRR * VV, BRR * HH);
}
